// RingAttention_37718402793788
// MI455X (gfx1250) — matrix operand order repaired and checked by reading, not compiled, not run on hardware
//
#include <hip/hip_runtime.h>

typedef __attribute__((ext_vector_type(16))) __bf16 v16bf;
typedef __attribute__((ext_vector_type(8)))  __bf16 v8bf;
typedef __attribute__((ext_vector_type(8)))  float  v8f;

constexpr int B_   = 32;
constexpr int SQ_  = 1024;
constexpr int SKV_ = 8192;
constexpr int D_   = 64;
constexpr int TKV  = 64;            // KV rows per tile
constexpr int NT_  = SKV_ / TKV;    // 128 tiles per batch
constexpr int WAVES= 4;
constexpr int QW   = 16;
constexpr int TILE_ELEMS = TKV * D_;                        // 4096 bf16 = 8 KB
constexpr size_t WS_HALF = (size_t)B_ * NT_ * TILE_ELEMS;   // elems per matrix
constexpr size_t WS_NEED = 2 * WS_HALF * sizeof(short);     // ~67 MB

// Pack two f32 into a bf16x2 dword with ONE v_perm_b32 (truncating bf16).
__device__ __forceinline__ unsigned pk_bf16(float a, float b) {
    return __builtin_amdgcn_perm(__builtin_bit_cast(unsigned, b),
                                 __builtin_bit_cast(unsigned, a),
                                 0x07060302u);
}

// One xor-butterfly step across the 16-lane row: v_permlane16_b32 + v_max.
__device__ __forceinline__ void pmax16(float& v, unsigned selLo, unsigned selHi) {
    float t;
    asm("v_permlane16_b32 %0, %1, %2, %3\n\t"
        "v_max_num_f32 %1, %1, %0"
        : "=&v"(t), "+v"(v)
        : "s"(selLo), "s"(selHi));
}

__device__ __forceinline__ float fmax_raw(float a, float b) {
    float d;
    asm("v_max_num_f32 %0, %1, %2" : "=v"(d) : "v"(a), "v"(b));
    return d;
}

// Async global->LDS copy of 64 B per lane (4 x b128), ASYNCcnt-tracked.
// GVS mode: global = SGPR base + 32-bit lane offset + imm; imm also offsets LDS.
__device__ __forceinline__ void async_copy64(unsigned ldsAddr, unsigned vOff,
                                             const void* sBase) {
    asm volatile(
        "global_load_async_to_lds_b128 %0, %1, %2 offset:0\n\t"
        "global_load_async_to_lds_b128 %0, %1, %2 offset:16\n\t"
        "global_load_async_to_lds_b128 %0, %1, %2 offset:32\n\t"
        "global_load_async_to_lds_b128 %0, %1, %2 offset:48"
        :: "v"(ldsAddr), "v"(vOff), "s"(sBase) : "memory");
}

// ---------------------------------------------------------------------------
// Pre-pass: K -> bf16 transposed tiles [d][kv] (kv dword-paired), V -> bf16
// permuted-row tiles; each tile contiguous (8 KB). Done once per call.
// ---------------------------------------------------------------------------
__global__ __launch_bounds__(128)
void prep_kv(const float* __restrict__ Kg, const float* __restrict__ Vg,
             __bf16* __restrict__ wsK, __bf16* __restrict__ wsV)
{
    __shared__ __attribute__((aligned(32))) __bf16 sKt[D_][TKV];
    __shared__ __attribute__((aligned(32))) __bf16 sV [TKV][D_];

    const int tid = threadIdx.x;
    const int b   = blockIdx.x >> 7;
    const int t   = blockIdx.x & (NT_ - 1);
    const int kv0 = t * TKV;

    const float* Kb = Kg + (size_t)b * SKV_ * D_;
    const float* Vb = Vg + (size_t)b * SKV_ * D_;

    {
        const int p  = tid & 31;
        const int g  = tid >> 5;
        const int d0 = 16 * g;
        const float* kr0 = Kb + (size_t)(kv0 + 2 * p) * D_ + d0;
        const float* kr1 = kr0 + D_;
        float ea[16], eb[16];
#pragma unroll
        for (int j = 0; j < 4; ++j) {
            float4 a = ((const float4*)kr0)[j];
            float4 bq = ((const float4*)kr1)[j];
            ea[4*j+0]=a.x;  ea[4*j+1]=a.y;  ea[4*j+2]=a.z;  ea[4*j+3]=a.w;
            eb[4*j+0]=bq.x; eb[4*j+1]=bq.y; eb[4*j+2]=bq.z; eb[4*j+3]=bq.w;
        }
#pragma unroll
        for (int i = 0; i < 16; ++i)
            *(unsigned*)&sKt[d0 + i][2 * p] = pk_bf16(ea[i], eb[i]);
    }
    {
        const int r    = tid >> 1;
        const int c0   = (tid & 1) * 32;
        const int rl   = r & 31;
        const int slot = 32 * (r >> 5) + ((rl & 15) << 1) + (rl >> 4);
        const float4* vp = (const float4*)(Vb + (size_t)(kv0 + r) * D_ + c0);
#pragma unroll
        for (int j = 0; j < 2; ++j) {
            float4 v0 = vp[4*j+0], v1 = vp[4*j+1], v2 = vp[4*j+2], v3 = vp[4*j+3];
            uint4 w;
            w.x = pk_bf16(v0.x, v0.y);  w.y = pk_bf16(v0.z, v0.w);
            w.z = pk_bf16(v1.x, v1.y);  w.w = pk_bf16(v1.z, v1.w);
            *(uint4*)&sV[slot][c0 + 16*j] = w;
            uint4 x;
            x.x = pk_bf16(v2.x, v2.y);  x.y = pk_bf16(v2.z, v2.w);
            x.z = pk_bf16(v3.x, v3.y);  x.w = pk_bf16(v3.z, v3.w);
            *(uint4*)&sV[slot][c0 + 16*j + 8] = x;
        }
    }
    __syncthreads();

    const size_t tile = (size_t)(b * NT_ + t) * TILE_ELEMS;
    {
        const __bf16* fK = &sKt[0][0] + tid * 32;
        const __bf16* fV = &sV [0][0] + tid * 32;
        __bf16* dK = wsK + tile + tid * 32;
        __bf16* dV = wsV + tile + tid * 32;
#pragma unroll
        for (int j = 0; j < 4; ++j) {
            ((uint4*)dK)[j] = ((const uint4*)fK)[j];
            ((uint4*)dV)[j] = ((const uint4*)fV)[j];
        }
    }
}

// ---------------------------------------------------------------------------
// Main loop: double-buffered async global->LDS staging of pre-converted
// tiles; next tile's copy overlaps current tile's WMMAs + softmax.
// ---------------------------------------------------------------------------
__global__ __launch_bounds__(WAVES * 32)
void fa_fwd_pre(const float* __restrict__ Qg, const __bf16* __restrict__ wsK,
                const __bf16* __restrict__ wsV, float* __restrict__ Og)
{
    __shared__ __attribute__((aligned(32))) __bf16 sK2[2][D_][TKV];     // 16 KB
    __shared__ __attribute__((aligned(32))) __bf16 sV2[2][TKV][D_];     // 16 KB
    __shared__ __attribute__((aligned(32))) __bf16 sP [WAVES][QW][TKV]; //  8 KB

    const int tid  = threadIdx.x;
    const int wave = tid >> 5;
    const int lane = tid & 31;

    const int b    = blockIdx.x >> 4;
    const int qblk = blockIdx.x & 15;
    const int q0   = qblk * (QW * WAVES) + wave * QW;

    const float* Qb = Qg + (size_t)b * SQ_ * D_;
    float*       Ob = Og + (size_t)b * SQ_ * D_;

    const float LOG2E = 1.44269504088896340736f;
    v16bf qa[2];
    {
        const int   qrow  = q0 + (lane & 15);
        const int   dbase = (lane < 16) ? 0 : 8;
        const float* qp   = Qb + (size_t)qrow * D_;
#pragma unroll
        for (int h = 0; h < 2; ++h) {
            union { unsigned u[8]; v16bf v; } qu;
#pragma unroll
            for (int j = 0; j < 8; ++j) {
                const int s0 = 2 * j, s1 = 2 * j + 1;
                const float f0 = qp[h * 32 + dbase + s0 + (s0 >= 8 ? 8 : 0)] * LOG2E;
                const float f1 = qp[h * 32 + dbase + s1 + (s1 >= 8 ? 8 : 0)] * LOG2E;
                qu.u[j] = pk_bf16(f0, f1);
            }
            qa[h] = qu.v;
        }
    }

    v16bf vones;
#pragma unroll
    for (int i = 0; i < 16; ++i) vones[i] = (__bf16)1.0f;

    float m_r[8], l_r[8];
    v8f   acc[4];
#pragma unroll
    for (int r = 0; r < 8; ++r) { m_r[r] = -__builtin_inff(); l_r[r] = 0.0f; }
#pragma unroll
    for (int t = 0; t < 4; ++t)
#pragma unroll
        for (int r = 0; r < 8; ++r) acc[t][r] = 0.0f;

    const unsigned ldsK = (unsigned)(size_t)&sK2[0][0][0];
    const unsigned ldsV = (unsigned)(size_t)&sV2[0][0][0];
    const unsigned thrOff = (unsigned)tid * 64u;

    // prologue: stage tile 0 into buffer 0
    {
        const __bf16* gK = wsK + (size_t)(b * NT_ + 0) * TILE_ELEMS;
        const __bf16* gV = wsV + (size_t)(b * NT_ + 0) * TILE_ELEMS;
        async_copy64(ldsK + thrOff, thrOff, gK);
        async_copy64(ldsV + thrOff, thrOff, gV);
        asm volatile("s_wait_asynccnt 0" ::: "memory");
    }
    __syncthreads();

    for (int t = 0; t < NT_; ++t) {
        const int cur = t & 1;
        // issue next tile's async copy into the other buffer (overlaps compute)
        {
            const int tn = (t + 1 < NT_) ? t + 1 : t;
            const int nb = cur ^ 1;
            const __bf16* gK = wsK + (size_t)(b * NT_ + tn) * TILE_ELEMS;
            const __bf16* gV = wsV + (size_t)(b * NT_ + tn) * TILE_ELEMS;
            async_copy64(ldsK + nb * 8192u + thrOff, thrOff, gK);
            async_copy64(ldsV + nb * 8192u + thrOff, thrOff, gV);
        }

        // ---- S = (Q*log2e) K^T ----
        v8f s[4];
#pragma unroll
        for (int nt = 0; nt < 4; ++nt) {
            v8f c;
#pragma unroll
            for (int r = 0; r < 8; ++r) c[r] = 0.0f;
#pragma unroll
            for (int dh = 0; dh < 2; ++dh) {
                v16bf kb;
#pragma unroll
                for (int s = 0; s < 16; ++s)
                    kb[s] = sK2[cur][dh * 32 + (lane >> 4) * 8 + (s & 7) + (s >> 3) * 16][nt * 16 + (lane & 15)];
                c = __builtin_amdgcn_wmma_f32_16x16x32_bf16(false, qa[dh], false, kb,
                                                            (short)0, c, false, false);
            }
            s[nt] = c;
        }

        // ---- row max ----
        float mn[8], alpha[8];
#pragma unroll
        for (int r = 0; r < 8; ++r) {
            float v = fmax_raw(fmax_raw(s[0][r], s[1][r]), fmax_raw(s[2][r], s[3][r]));
            pmax16(v, 0x67452301u, 0xEFCDAB89u);
            pmax16(v, 0x54761032u, 0xDCFE98BAu);
            pmax16(v, 0x32107654u, 0xBA98FEDCu);
            pmax16(v, 0xFEDCBA98u, 0x76543210u);
            mn[r]    = fmax_raw(m_r[r], v);
            alpha[r] = __builtin_amdgcn_exp2f(m_r[r] - mn[r]);
            m_r[r]   = mn[r];
        }
#pragma unroll
        for (int nt = 0; nt < 4; ++nt)
#pragma unroll
            for (int r = 0; r < 8; ++r)
                s[nt][r] = __builtin_amdgcn_exp2f(s[nt][r] - mn[r]);

        // ---- P -> per-wave scratch (interleaved columns) ----
        {
            const int qrow = (lane >> 4) * 8;
            const int n    = lane & 15;
#pragma unroll
            for (int r = 0; r < 8; ++r) {
                *(unsigned*)&sP[wave][qrow + r][2 * n]      = pk_bf16(s[0][r], s[1][r]);
                *(unsigned*)&sP[wave][qrow + r][32 + 2 * n] = pk_bf16(s[2][r], s[3][r]);
            }
        }
        asm volatile("s_wait_dscnt 0" ::: "memory");
        v16bf pa[2];
        {
            const int q     = lane & 15;
            const int kbase = (lane < 16) ? 0 : 8;
#pragma unroll
            for (int hh = 0; hh < 2; ++hh) {
                const __bf16* row = &sP[wave][q][32 * hh];
                v8bf p0 = *(const v8bf*)(row + kbase);
                v8bf p1 = *(const v8bf*)(row + kbase + 16);
                pa[hh] = __builtin_shufflevector(p0, p1,
                         0,1,2,3,4,5,6,7,8,9,10,11,12,13,14,15);
            }
        }

        // ---- row sums via WMMA against ones ----
        {
            v8f csum;
#pragma unroll
            for (int r = 0; r < 8; ++r) csum[r] = 0.0f;
            csum = __builtin_amdgcn_wmma_f32_16x16x32_bf16(false, pa[0], false, vones,
                                                           (short)0, csum, false, false);
            csum = __builtin_amdgcn_wmma_f32_16x16x32_bf16(false, pa[1], false, vones,
                                                           (short)0, csum, false, false);
#pragma unroll
            for (int r = 0; r < 8; ++r) l_r[r] = l_r[r] * alpha[r] + csum[r];
        }

#pragma unroll
        for (int tt = 0; tt < 4; ++tt)
#pragma unroll
            for (int r = 0; r < 8; ++r) acc[tt][r] *= alpha[r];

#pragma unroll
        for (int tt = 0; tt < 4; ++tt) {
            v16bf vb0, vb1;
#pragma unroll
            for (int s = 0; s < 16; ++s) {
                const int kr = (lane >> 4) * 8 + (s & 7) + (s >> 3) * 16;
                vb0[s] = sV2[cur][kr][tt * 16 + (lane & 15)];
                vb1[s] = sV2[cur][32 + kr][tt * 16 + (lane & 15)];
            }
            acc[tt] = __builtin_amdgcn_wmma_f32_16x16x32_bf16(false, pa[0], false, vb0,
                                                              (short)0, acc[tt], false, false);
            acc[tt] = __builtin_amdgcn_wmma_f32_16x16x32_bf16(false, pa[1], false, vb1,
                                                              (short)0, acc[tt], false, false);
        }

        // next tile fully in LDS + everyone done with current buffers
        asm volatile("s_wait_asynccnt 0" ::: "memory");
        __syncthreads();
    }

    {
        const int qrow = (lane >> 4) * 8;
        const int n    = lane & 15;
#pragma unroll
        for (int r = 0; r < 8; ++r) {
            const float inv = 1.0f / l_r[r];
            float* op = Ob + (size_t)(q0 + qrow + r) * D_ + n;
#pragma unroll
            for (int tt = 0; tt < 4; ++tt) op[tt * 16] = acc[tt][r] * inv;
        }
    }
}

// ---------------------------------------------------------------------------
// Fallback (self-contained) used only if d_ws is too small.
// ---------------------------------------------------------------------------
__global__ __launch_bounds__(WAVES * 32)
void fa_fwd_bf16_wmma(const float* __restrict__ Qg, const float* __restrict__ Kg,
                      const float* __restrict__ Vg, float* __restrict__ Og)
{
    __shared__ __attribute__((aligned(32))) __bf16 sKt[D_][TKV];
    __shared__ __attribute__((aligned(32))) __bf16 sV [TKV][D_];
    __shared__ __attribute__((aligned(32))) __bf16 sP [WAVES][QW][TKV];

    const int tid  = threadIdx.x;
    const int wave = tid >> 5;
    const int lane = tid & 31;
    const int b    = blockIdx.x >> 4;
    const int qblk = blockIdx.x & 15;
    const int q0   = qblk * (QW * WAVES) + wave * QW;

    const float* Qb = Qg + (size_t)b * SQ_  * D_;
    const float* Kb = Kg + (size_t)b * SKV_ * D_;
    const float* Vb = Vg + (size_t)b * SKV_ * D_;
    float*       Ob = Og + (size_t)b * SQ_  * D_;

    const float LOG2E = 1.44269504088896340736f;
    v16bf qa[2];
    {
        const int   qrow  = q0 + (lane & 15);
        const int   dbase = (lane < 16) ? 0 : 8;
        const float* qp   = Qb + (size_t)qrow * D_;
#pragma unroll
        for (int h = 0; h < 2; ++h) {
            union { unsigned u[8]; v16bf v; } qu;
#pragma unroll
            for (int j = 0; j < 8; ++j) {
                const int s0 = 2 * j, s1 = 2 * j + 1;
                const float f0 = qp[h * 32 + dbase + s0 + (s0 >= 8 ? 8 : 0)] * LOG2E;
                const float f1 = qp[h * 32 + dbase + s1 + (s1 >= 8 ? 8 : 0)] * LOG2E;
                qu.u[j] = pk_bf16(f0, f1);
            }
            qa[h] = qu.v;
        }
    }
    v16bf vones;
#pragma unroll
    for (int i = 0; i < 16; ++i) vones[i] = (__bf16)1.0f;

    float m_r[8], l_r[8];
    v8f   acc[4];
#pragma unroll
    for (int r = 0; r < 8; ++r) { m_r[r] = -__builtin_inff(); l_r[r] = 0.0f; }
#pragma unroll
    for (int t = 0; t < 4; ++t)
#pragma unroll
        for (int r = 0; r < 8; ++r) acc[t][r] = 0.0f;

    for (int kv0 = 0; kv0 < SKV_; kv0 += TKV) {
        __syncthreads();
        {
            const int p  = tid & 31;
            const int g  = tid >> 5;
            const int d0 = 16 * g;
            const float* kr0 = Kb + (size_t)(kv0 + 2 * p) * D_ + d0;
            const float* kr1 = kr0 + D_;
            float ea[16], eb[16];
#pragma unroll
            for (int j = 0; j < 4; ++j) {
                float4 a = ((const float4*)kr0)[j];
                float4 bq = ((const float4*)kr1)[j];
                ea[4*j+0]=a.x;  ea[4*j+1]=a.y;  ea[4*j+2]=a.z;  ea[4*j+3]=a.w;
                eb[4*j+0]=bq.x; eb[4*j+1]=bq.y; eb[4*j+2]=bq.z; eb[4*j+3]=bq.w;
            }
#pragma unroll
            for (int i = 0; i < 16; ++i)
                *(unsigned*)&sKt[d0 + i][2 * p] = pk_bf16(ea[i], eb[i]);
        }
        {
            const int r    = tid >> 1;
            const int c0   = (tid & 1) * 32;
            const int rl   = r & 31;
            const int slot = 32 * (r >> 5) + ((rl & 15) << 1) + (rl >> 4);
            const float4* vp = (const float4*)(Vb + (size_t)(kv0 + r) * D_ + c0);
#pragma unroll
            for (int j = 0; j < 2; ++j) {
                float4 v0 = vp[4*j+0], v1 = vp[4*j+1], v2 = vp[4*j+2], v3 = vp[4*j+3];
                uint4 w;
                w.x = pk_bf16(v0.x, v0.y);  w.y = pk_bf16(v0.z, v0.w);
                w.z = pk_bf16(v1.x, v1.y);  w.w = pk_bf16(v1.z, v1.w);
                *(uint4*)&sV[slot][c0 + 16*j] = w;
                uint4 x;
                x.x = pk_bf16(v2.x, v2.y);  x.y = pk_bf16(v2.z, v2.w);
                x.z = pk_bf16(v3.x, v3.y);  x.w = pk_bf16(v3.z, v3.w);
                *(uint4*)&sV[slot][c0 + 16*j + 8] = x;
            }
        }
        __syncthreads();

        v8f s[4];
#pragma unroll
        for (int nt = 0; nt < 4; ++nt) {
            v8f c;
#pragma unroll
            for (int r = 0; r < 8; ++r) c[r] = 0.0f;
#pragma unroll
            for (int dh = 0; dh < 2; ++dh) {
                v16bf kb;
#pragma unroll
                for (int s = 0; s < 16; ++s)
                    kb[s] = sKt[dh * 32 + (lane >> 4) * 8 + (s & 7) + (s >> 3) * 16][nt * 16 + (lane & 15)];
                c = __builtin_amdgcn_wmma_f32_16x16x32_bf16(false, qa[dh], false, kb,
                                                            (short)0, c, false, false);
            }
            s[nt] = c;
        }

        float mn[8], alpha[8];
#pragma unroll
        for (int r = 0; r < 8; ++r) {
            float v = fmax_raw(fmax_raw(s[0][r], s[1][r]), fmax_raw(s[2][r], s[3][r]));
            pmax16(v, 0x67452301u, 0xEFCDAB89u);
            pmax16(v, 0x54761032u, 0xDCFE98BAu);
            pmax16(v, 0x32107654u, 0xBA98FEDCu);
            pmax16(v, 0xFEDCBA98u, 0x76543210u);
            mn[r]    = fmax_raw(m_r[r], v);
            alpha[r] = __builtin_amdgcn_exp2f(m_r[r] - mn[r]);
            m_r[r]   = mn[r];
        }
#pragma unroll
        for (int nt = 0; nt < 4; ++nt)
#pragma unroll
            for (int r = 0; r < 8; ++r)
                s[nt][r] = __builtin_amdgcn_exp2f(s[nt][r] - mn[r]);

        {
            const int qrow = (lane >> 4) * 8;
            const int n    = lane & 15;
#pragma unroll
            for (int r = 0; r < 8; ++r) {
                *(unsigned*)&sP[wave][qrow + r][2 * n]      = pk_bf16(s[0][r], s[1][r]);
                *(unsigned*)&sP[wave][qrow + r][32 + 2 * n] = pk_bf16(s[2][r], s[3][r]);
            }
        }
        asm volatile("s_wait_dscnt 0" ::: "memory");
        v16bf pa[2];
        {
            const int q     = lane & 15;
            const int kbase = (lane < 16) ? 0 : 8;
#pragma unroll
            for (int hh = 0; hh < 2; ++hh) {
                const __bf16* row = &sP[wave][q][32 * hh];
                v8bf p0 = *(const v8bf*)(row + kbase);
                v8bf p1 = *(const v8bf*)(row + kbase + 16);
                pa[hh] = __builtin_shufflevector(p0, p1,
                         0,1,2,3,4,5,6,7,8,9,10,11,12,13,14,15);
            }
        }
        {
            v8f csum;
#pragma unroll
            for (int r = 0; r < 8; ++r) csum[r] = 0.0f;
            csum = __builtin_amdgcn_wmma_f32_16x16x32_bf16(false, pa[0], false, vones,
                                                           (short)0, csum, false, false);
            csum = __builtin_amdgcn_wmma_f32_16x16x32_bf16(false, pa[1], false, vones,
                                                           (short)0, csum, false, false);
#pragma unroll
            for (int r = 0; r < 8; ++r) l_r[r] = l_r[r] * alpha[r] + csum[r];
        }
#pragma unroll
        for (int t = 0; t < 4; ++t)
#pragma unroll
            for (int r = 0; r < 8; ++r) acc[t][r] *= alpha[r];
#pragma unroll
        for (int t = 0; t < 4; ++t) {
            v16bf vb0, vb1;
#pragma unroll
            for (int s = 0; s < 16; ++s) {
                const int kr = (lane >> 4) * 8 + (s & 7) + (s >> 3) * 16;
                vb0[s] = sV[kr][t * 16 + (lane & 15)];
                vb1[s] = sV[32 + kr][t * 16 + (lane & 15)];
            }
            acc[t] = __builtin_amdgcn_wmma_f32_16x16x32_bf16(false, pa[0], false, vb0,
                                                             (short)0, acc[t], false, false);
            acc[t] = __builtin_amdgcn_wmma_f32_16x16x32_bf16(false, pa[1], false, vb1,
                                                             (short)0, acc[t], false, false);
        }
    }
    {
        const int qrow = (lane >> 4) * 8;
        const int n    = lane & 15;
#pragma unroll
        for (int r = 0; r < 8; ++r) {
            const float inv = 1.0f / l_r[r];
            float* op = Ob + (size_t)(q0 + qrow + r) * D_ + n;
#pragma unroll
            for (int t = 0; t < 4; ++t) op[t * 16] = acc[t][r] * inv;
        }
    }
}

extern "C" void kernel_launch(void* const* d_in, const int* in_sizes, int n_in,
                              void* d_out, int out_size, void* d_ws, size_t ws_size,
                              hipStream_t stream)
{
    (void)in_sizes; (void)n_in; (void)out_size;
    const float* Q = (const float*)d_in[0];
    const float* K = (const float*)d_in[1];
    const float* V = (const float*)d_in[2];
    float*       O = (float*)d_out;

    dim3 grid(B_ * (SQ_ / (QW * WAVES)));   // 512 workgroups
    dim3 block(WAVES * 32);                 // 4 wave32

    if (ws_size >= WS_NEED && d_ws != nullptr) {
        __bf16* wsK = (__bf16*)d_ws;
        __bf16* wsV = wsK + WS_HALF;
        prep_kv<<<dim3(B_ * NT_), block, 0, stream>>>(K, V, wsK, wsV);
        fa_fwd_pre<<<grid, block, 0, stream>>>(Q, wsK, wsV, O);
    } else {
        fa_fwd_bf16_wmma<<<grid, block, 0, stream>>>(Q, K, V, O);
    }
}
